// EnhancedTransformerEncoder_64931315581524
// MI455X (gfx1250) — matrix operand order repaired and checked by reading, not compiled, not run on hardware
//
#include <hip/hip_runtime.h>
#include <hip/hip_bf16.h>
#include <math.h>

// ---------------- problem constants ----------------
#define TT 128
#define BB 16
#define DD 512
#define HH 8
#define HDQ 64
#define ETA 4
#define RR 8
#define FF (ETA * HDQ)          // 256
#define MROWS (TT * BB)         // 2048
#define NCAT (5 * DD + 3 * (HH * ETA))  // 2656
#define EPSV 1e-6f

typedef __attribute__((ext_vector_type(16))) _Float16 v16h;
typedef __attribute__((ext_vector_type(8)))  _Float16 v8h;
typedef __attribute__((ext_vector_type(8)))  float    v8f;

// ---------------- fp32 -> fp16 conversion ----------------
__global__ void cvt_f32_to_f16(const float* __restrict__ src,
                               _Float16* __restrict__ dst, int n) {
  int i = blockIdx.x * blockDim.x + threadIdx.x;
  if (i < n) dst[i] = (_Float16)src[i];
}

// ---------------- WMMA GEMM: C[M,N] = A[M,K] * B[N,K]^T (+bias[N]) ---------
// One wave per 64x32 C block: 4 M-tiles x 2 N-tiles of 16x16, register-tiled
// so each A fragment feeds 2 WMMAs and each B fragment feeds 4 WMMAs
// (~21 flops per byte of L2 traffic vs 8 for the untiled version).
// Lane layouts (A/B in this code's contiguous k order; C/D per CDNA5 ISA 7.12.2):
//   A (16x32 f16): lane={m,hi}; elems 0..15 = K[hi*16..+15]
//   B (32x16 f16): lane={n,hi}; elems 0..15 = K[hi*16..+15]
//   C/D (16x16 f32): elem i -> row = i + 8*hi, col = lane&15
__global__ __launch_bounds__(32)
void wmma_gemm_f16(const _Float16* __restrict__ A,   // M x K row-major
                   const _Float16* __restrict__ Bw,  // N x K row-major (weights)
                   const float* __restrict__ bias,   // N or nullptr
                   float* __restrict__ C,            // M x N row-major
                   int M, int N, int K) {
  const int nb = blockIdx.x;       // N block of 32
  const int mb = blockIdx.y;       // M block of 64
  const int lane = threadIdx.x;
  const int lo = lane & 15;
  const int hi = lane >> 4;
  const int n0 = nb * 32, m0 = mb * 64;

  v8f c[4][2];
#pragma unroll
  for (int j = 0; j < 2; ++j) {
    const float bv = bias ? bias[n0 + j * 16 + lo] : 0.0f;
#pragma unroll
    for (int i = 0; i < 4; ++i)
#pragma unroll
      for (int q = 0; q < 8; ++q) c[i][j][q] = bv;
  }

  const _Float16* arow[4];
#pragma unroll
  for (int i = 0; i < 4; ++i) arow[i] = A + (size_t)(m0 + i * 16 + lo) * K;
  const _Float16* brow[2];
#pragma unroll
  for (int j = 0; j < 2; ++j) brow[j] = Bw + (size_t)(n0 + j * 16 + lo) * K;

  for (int k0 = 0; k0 < K; k0 += 32) {
    if (k0 + 128 < K) {  // pull next k-slices toward the WGP (global_prefetch_b8)
      __builtin_prefetch((const void*)(arow[0] + k0 + 128));
      __builtin_prefetch((const void*)(brow[0] + k0 + 128));
    }
    v16h a[4];
#pragma unroll
    for (int i = 0; i < 4; ++i) {
      a[i] = *(const v16h*)(arow[i] + k0 + hi * 16);
    }
    v16h b[2];
#pragma unroll
    for (int j = 0; j < 2; ++j)
      b[j] = *(const v16h*)(brow[j] + k0 + hi * 16);

#pragma unroll
    for (int i = 0; i < 4; ++i)
#pragma unroll
      for (int j = 0; j < 2; ++j)
        c[i][j] = __builtin_amdgcn_wmma_f32_16x16x32_f16(
            false, a[i], false, b[j], (short)0, c[i][j], false, false);
  }

#pragma unroll
  for (int i = 0; i < 4; ++i) {
    const int rbase = m0 + i * 16 + hi * 8;
#pragma unroll
    for (int j = 0; j < 2; ++j) {
      const int col = n0 + j * 16 + lo;
#pragma unroll
      for (int q = 0; q < 8; ++q)
        C[(size_t)(rbase + q) * N + col] = c[i][j][q];
    }
  }
}

// ---------------- fused scan + attention contraction ----------------
__device__ __forceinline__ float sigmoidf_(float x) {
  return 1.0f / (1.0f + __expf(-x));
}

__device__ __forceinline__ float wave_red32(float v) {
#pragma unroll
  for (int off = 16; off; off >>= 1) v += __shfl_xor(v, off, 32);
  return v;
}

// grid = B*H blocks, 256 threads. Thread tid owns feature f = tid
// (e = tid>>6, d = tid&63). State kept in registers across the T scan:
//   s[f], tilde_k[r][f] (8 regs), tilde_v pair (r0 = tid>>6, r0+4).
// All 9 per-step reductions (kdq[0..7], norm) are fused: wave32 shfl
// butterflies + one LDS stage + ONE barrier (3 barriers per timestep total).
__global__ __launch_bounds__(256)
void scan_attn_kernel(const float* __restrict__ P,        // 2048 x NCAT
                      const int*   __restrict__ term,     // T x B
                      const float* __restrict__ tk_prev,  // B,R,H,F
                      const float* __restrict__ tv_prev,  // B,R,H,HD
                      const float* __restrict__ s_prev,   // B,H,F
                      const float* __restrict__ tick,     // B
                      const float* __restrict__ omegas,   // R
                      float* __restrict__ attn,           // 2048 x 512
                      float* __restrict__ out_tk,
                      float* __restrict__ out_tv,
                      float* __restrict__ out_s,
                      float* __restrict__ out_tick) {
  const int bh = blockIdx.x;
  const int b = bh >> 3;
  const int h = bh & 7;
  const int tid = threadIdx.x;
  const int e = tid >> 6;        // 0..3  (eta index)
  const int d = tid & 63;        // 0..63 (head-dim index)
  const int r0 = tid >> 6;       // tv rows handled: r0 and r0+4
  const int wave = tid >> 5;
  const int lane = tid & 31;

  __shared__ float st[336];      // q[0:64) k[64:128) v[128:192) beta[192:256)
                                 // gamma[256:320) p1[320:324) p2[324:328) p3[328:332)
  __shared__ float wpart[8][9];  // per-wave partials: kdq[0..7], norm
  __shared__ float red[256];

  float s_state = s_prev[((size_t)(b * HH + h)) * FF + tid];
  float tk[RR];
#pragma unroll
  for (int r = 0; r < RR; ++r)
    tk[r] = tk_prev[((size_t)((b * RR + r) * HH + h)) * FF + tid];
  float tv0 = tv_prev[((size_t)((b * RR + r0) * HH + h)) * HDQ + d];
  float tv1 = tv_prev[((size_t)((b * RR + r0 + 4) * HH + h)) * HDQ + d];

  float om[RR];
#pragma unroll
  for (int r = 0; r < RR; ++r) om[r] = omegas[r];
  const float om_lo = om[r0], om_hi = om[r0 + 4];
  const float tick0 = tick[b];

  for (int t = 0; t < TT; ++t) {
    const float* row = P + (size_t)(t * BB + b) * NCAT;
    if (tid < 64) {
      st[tid]       = row[0 * DD + h * HDQ + tid];   // q
      st[64 + tid]  = row[1 * DD + h * HDQ + tid];   // k
      st[128 + tid] = row[2 * DD + h * HDQ + tid];   // v
      st[192 + tid] = row[3 * DD + h * HDQ + tid];   // beta pre-act
      st[256 + tid] = row[4 * DD + h * HDQ + tid];   // gamma pre-act
    }
    if (tid < 4) {
      st[320 + tid] = row[5 * DD + 0 * HH * ETA + h * ETA + tid];  // p1
      st[324 + tid] = row[5 * DD + 1 * HH * ETA + h * ETA + tid];  // p2
      st[328 + tid] = row[5 * DD + 2 * HH * ETA + h * ETA + tid];  // p3
    }
    __syncthreads();                                       // barrier 1

    const float mask = 1.0f - (float)term[t * BB + b];

    const float relu_q   = fmaxf(st[d], 0.0f);
    const float relu_k   = fmaxf(st[64 + d], 0.0f);
    const float sig_beta = sigmoidf_(st[192 + d]);
    const float sig_gam  = sigmoidf_(st[256 + d]);
    const float p1r = fmaxf(st[320 + e], 0.0f);
    const float p2r = fmaxf(st[324 + e], 0.0f);
    const float p3s = sigmoidf_(st[328 + e]);

    const float phi_q = p2r * relu_q;                 // feature map of q
    const float gfeat = p3s * sig_gam;                // gamma feature
    const float gk = (p1r * relu_k) * gfeat;          // gated_keys[f]
    const float dg = (1.0f - gfeat) * mask;           // discount_gamma[f]
    const float gv = st[128 + d] * sig_beta;          // gated_values[d]
    const float db = (1.0f - sig_beta) * mask;        // discount_beta[d]

    const float tt = (float)(t + 1) + tick0;

    // scan updates (state = disc*state + x); post-update state used below
    s_state = dg * s_state + gk;
#pragma unroll
    for (int r = 0; r < RR; ++r)
      tk[r] = dg * tk[r] + gk * __cosf(tt * om[r]);
    tv0 = db * tv0 + gv * __cosf(tt * om_lo);
    tv1 = db * tv1 + gv * __cosf(tt * om_hi);

    // fused reductions: kdq[r] = sum_f tk[r]*phi_q ; norm = sum_f s*phi_q
#pragma unroll
    for (int r = 0; r < RR; ++r) {
      float v = wave_red32(tk[r] * phi_q);
      if (lane == 0) wpart[wave][r] = v;
    }
    {
      float v = wave_red32(s_state * phi_q);
      if (lane == 0) wpart[wave][8] = v;
    }
    __syncthreads();                                       // barrier 2

    float norm = 0.f, kq0 = 0.f, kq1 = 0.f;
#pragma unroll
    for (int w = 0; w < 8; ++w) {
      norm += wpart[w][8];
      kq0  += wpart[w][r0];
      kq1  += wpart[w][r0 + 4];
    }

    red[tid] = tv0 * kq0 + tv1 * kq1;
    __syncthreads();                                       // barrier 3
    if (tid < 64) {
      float kv = red[tid] + red[tid + 64] + red[tid + 128] + red[tid + 192];
      attn[(size_t)(t * BB + b) * (HH * HDQ) + h * HDQ + tid] =
          kv / (2.0f * (float)RR * norm + EPSV);
    }
    // no end barrier needed: next write to st/wpart/red is separated from all
    // readers by the barriers above in the following iteration
  }

  // final states
  out_s[((size_t)(b * HH + h)) * FF + tid] = s_state;
#pragma unroll
  for (int r = 0; r < RR; ++r)
    out_tk[((size_t)((b * RR + r) * HH + h)) * FF + tid] = tk[r];
  out_tv[((size_t)((b * RR + r0) * HH + h)) * HDQ + d] = tv0;
  out_tv[((size_t)((b * RR + r0 + 4) * HH + h)) * HDQ + d] = tv1;
  if (h == 0 && tid == 0) out_tick[b] = tick0 + (float)TT;
}

// ---------------- host-side launch ----------------
static inline size_t align256(size_t x) { return (x + 255) & ~(size_t)255; }

extern "C" void kernel_launch(void* const* d_in, const int* in_sizes, int n_in,
                              void* d_out, int out_size, void* d_ws, size_t ws_size,
                              hipStream_t stream) {
  const float* inputs  = (const float*)d_in[0];   // T,B,D
  const int*   term    = (const int*)  d_in[1];   // T,B
  const float* tk_prev = (const float*)d_in[2];   // B,R,H,F
  const float* tv_prev = (const float*)d_in[3];   // B,R,H,HD
  const float* s_prev  = (const float*)d_in[4];   // B,H,F
  const float* tick    = (const float*)d_in[5];   // B
  const float* Wq      = (const float*)d_in[6];
  const float* Wk      = (const float*)d_in[7];
  const float* Wv      = (const float*)d_in[8];
  const float* Wbeta   = (const float*)d_in[9];
  const float* Wgamma  = (const float*)d_in[10];
  const float* Wp1     = (const float*)d_in[11];
  const float* Wp2     = (const float*)d_in[12];
  const float* Wp3     = (const float*)d_in[13];
  const float* Wout    = (const float*)d_in[14];
  const float* bout    = (const float*)d_in[15];
  const float* omegas  = (const float*)d_in[16];

  // ---- workspace carve-up ----
  char* ws = (char*)d_ws;
  size_t off = 0;
  _Float16* Xh = (_Float16*)(ws + off); off += align256((size_t)MROWS * DD * 2);
  _Float16* Wcath = (_Float16*)(ws + off); off += align256((size_t)NCAT * DD * 2);
  _Float16* Wouth = (_Float16*)(ws + off); off += align256((size_t)DD * (HH * HDQ) * 2);
  float* P = (float*)(ws + off); off += align256((size_t)MROWS * NCAT * 4);
  float* attn = (float*)(ws + off); off += align256((size_t)MROWS * (HH * HDQ) * 4);
  _Float16* attnh = (_Float16*)(ws + off); off += align256((size_t)MROWS * (HH * HDQ) * 4);

  // ---- output carve-up (flat concat in return order) ----
  float* out_output = (float*)d_out;                       // T*B*D
  float* out_tk   = out_output + (size_t)MROWS * DD;       // B*R*H*F
  float* out_tv   = out_tk + (size_t)BB * RR * HH * FF;    // B*R*H*HD
  float* out_s    = out_tv + (size_t)BB * RR * HH * HDQ;   // B*H*F
  float* out_tick = out_s + (size_t)BB * HH * FF;          // B

  auto cvt = [&](const float* src, _Float16* dst, int n) {
    cvt_f32_to_f16<<<(n + 255) / 256, 256, 0, stream>>>(src, dst, n);
  };

  // activations + concatenated weights -> fp16
  cvt(inputs, Xh, MROWS * DD);
  cvt(Wq,     Wcath + (size_t)0    * DD, DD * DD);
  cvt(Wk,     Wcath + (size_t)512  * DD, DD * DD);
  cvt(Wv,     Wcath + (size_t)1024 * DD, DD * DD);
  cvt(Wbeta,  Wcath + (size_t)1536 * DD, DD * DD);
  cvt(Wgamma, Wcath + (size_t)2048 * DD, DD * DD);
  cvt(Wp1,    Wcath + (size_t)2560 * DD, HH * ETA * DD);
  cvt(Wp2,    Wcath + (size_t)2592 * DD, HH * ETA * DD);
  cvt(Wp3,    Wcath + (size_t)2624 * DD, HH * ETA * DD);
  cvt(Wout,   Wouth,                     DD * (HH * HDQ));

  // fused projection GEMM: P[2048,2656] = X * Wcat^T  (WMMA f16, 4x2 tiles)
  wmma_gemm_f16<<<dim3(NCAT / 32, MROWS / 64), 32, 0, stream>>>(
      Xh, Wcath, nullptr, P, MROWS, NCAT, DD);

  // fused scan + attention (keeps the 134MB scan tensor entirely in registers)
  scan_attn_kernel<<<BB * HH, 256, 0, stream>>>(
      P, term, tk_prev, tv_prev, s_prev, tick, omegas,
      attn, out_tk, out_tv, out_s, out_tick);

  // output GEMM: out = attn * Wout^T + bout  (WMMA f16, bias fused)
  cvt(attn, attnh, MROWS * (HH * HDQ));
  wmma_gemm_f16<<<dim3(DD / 32, MROWS / 64), 32, 0, stream>>>(
      attnh, Wouth, bout, out_output, MROWS, DD, DD);
}
